// EqvSelfAttention_89550068122333
// MI455X (gfx1250) — compile-verified, matrix operand order checked by reading, not run on hardware
//
#include <hip/hip_runtime.h>
#include <hip/hip_bf16.h>

// B=4, N=1024, D=256, H=8, DH=32
#define INF_F 1.0e38f

typedef __attribute__((ext_vector_type(16))) _Float16 v16h;
typedef __attribute__((ext_vector_type(8)))  _Float16 v8h;
typedef __attribute__((ext_vector_type(8)))  float    v8f;
typedef __attribute__((ext_vector_type(4)))  float    f32x4;
typedef __attribute__((ext_vector_type(4)))  unsigned u32x4;
typedef __attribute__((ext_vector_type(8)))  unsigned u32x8;

__device__ __forceinline__ v16h join8(v8h lo, v8h hi) {
    v16h r;
#pragma unroll
    for (int i = 0; i < 8; ++i) { r[i] = lo[i]; r[i + 8] = hi[i]; }
    return r;
}

// A-fragment from row-major f32 (8+8 consecutive floats, f32->f16)
__device__ __forceinline__ v16h loadA_f32(const float* p) {
    f32x4 a0 = *(const f32x4*)(p);
    f32x4 a1 = *(const f32x4*)(p + 4);
    f32x4 b0 = *(const f32x4*)(p + 16);
    f32x4 b1 = *(const f32x4*)(p + 20);
    v16h r;
#pragma unroll
    for (int i = 0; i < 4; ++i) {
        r[i]      = (_Float16)a0[i];
        r[i + 4]  = (_Float16)a1[i];
        r[i + 8]  = (_Float16)b0[i];
        r[i + 12] = (_Float16)b1[i];
    }
    return r;
}

__device__ __forceinline__ float halfmax16(float v) {
#pragma unroll
    for (int m = 1; m < 16; m <<= 1) v = fmaxf(v, __shfl_xor(v, m, 32));
    return v;
}
__device__ __forceinline__ float halfsum16(float v) {
#pragma unroll
    for (int m = 1; m < 16; m <<= 1) v += __shfl_xor(v, m, 32);
    return v;
}

// pair-bias: loc = sum_d wg2[d]*relu(x @ Wg1[:,d] + bg1[d]) + bg2
__device__ __forceinline__ float locbias(const float* x, const float* g,
                                         const float* b1, const float* w2, float b2) {
    float acc = b2;
#pragma unroll
    for (int d = 0; d < 3; ++d) {
        float h = fmaf(x[0], g[0 * 3 + d],
                  fmaf(x[1], g[1 * 3 + d],
                  fmaf(x[2], g[2 * 3 + d], b1[d])));
        acc = fmaf(fmaxf(h, 0.f), w2[d], acc);
    }
    return acc;
}

// LDS byte offset of a __shared__ object (generic -> addrspace(3) -> u32)
__device__ __forceinline__ unsigned lds_off(const void* p) {
    return (unsigned)(uintptr_t)(__attribute__((address_space(3))) const void*)p;
}

// ---------------------------------------------------------------------------
// Tensor Data Mover: async 2D tile load Global -> LDS.
//   tile = tile_y rows of tile_x f32 elements, row stride = stride_elems,
//   packed contiguously into LDS at lds_addr. Tracked by TENSORcnt.
// D# per CDNA5 ISA 8.3/8.4 (group0 128b, group1 256b; groups 2/3 = NULL).
// ---------------------------------------------------------------------------
__device__ __forceinline__ void tdm_load_2d_f32(unsigned lds_addr, const void* gsrc,
                                                unsigned tile_x, unsigned tile_y,
                                                unsigned tensor_dim0, unsigned tensor_dim1,
                                                unsigned stride_elems) {
    const unsigned long long ga = (unsigned long long)(uintptr_t)gsrc;
    u32x4 g0;
    g0[0] = 1u;                                               // count=1 (valid), user mode
    g0[1] = lds_addr;                                         // lds_addr[31:0]
    g0[2] = (unsigned)(ga & 0xffffffffu);                     // global_addr[31:0]
    g0[3] = (unsigned)((ga >> 32) & 0x1ffffffu) | (2u << 30); // global_addr[56:32] | type=2
    u32x8 g1;
    g1[0] = (2u << 16);                                       // wg_mask=0, data_size=2 (4B)
    g1[1] = (tensor_dim0 & 0xffffu) << 16;                    // tensor_dim0[15:0]
    g1[2] = ((tensor_dim0 >> 16) & 0xffffu) | ((tensor_dim1 & 0xffffu) << 16);
    g1[3] = ((tensor_dim1 >> 16) & 0xffffu) | ((tile_x & 0xffffu) << 16);  // tile_dim0
    g1[4] = (tile_y & 0xffffu);                               // tile_dim1, tile_dim2=0
    g1[5] = stride_elems;                                     // tensor_dim0_stride[31:0]
    g1[6] = 0u;                                               // stride[47:32], dim1_stride lo
    g1[7] = 0u;
    asm volatile("tensor_load_to_lds %0, %1" :: "s"(g0), "s"(g1) : "memory");
}

// ---------------------------------------------------------------------------
// Kernel 0: weight prep — W[k][n] f32 -> Wt[n][k] f16 so WMMA B-fragments
// become contiguous 16B loads.
// ---------------------------------------------------------------------------
__global__ __launch_bounds__(256) void wprep_kernel(
    const float* __restrict__ Wq, const float* __restrict__ Wk,
    const float* __restrict__ Wv, const float* __restrict__ Wo,
    _Float16* __restrict__ Wqt, _Float16* __restrict__ Wkt,
    _Float16* __restrict__ Wvt, _Float16* __restrict__ Wot)
{
    const int n = threadIdx.x;        // 0..255
    const int k = blockIdx.x;         // 0..255
    const size_t src = (size_t)k * 256 + n;
    const size_t dst = (size_t)n * 256 + k;
    Wqt[dst] = (_Float16)Wq[src];
    Wkt[dst] = (_Float16)Wk[src];
    Wvt[dst] = (_Float16)Wv[src];
    Wot[dst] = (_Float16)Wo[src];
}

// ---------------------------------------------------------------------------
// Kernel 1: Q/K/V projections. One wave -> one 16x16 output tile, 8 K-steps
// of v_wmma_f32_16x16x32_f16. Stores Q (scaled by 1/sqrt(D)) & K row-major
// f16, V as f32 (residual) and transposed f16 [b][h][d][n] for attention.
// ---------------------------------------------------------------------------
__global__ __launch_bounds__(32) void qkv_kernel(
    const float* __restrict__ Y,
    const _Float16* __restrict__ Wqt, const float* __restrict__ bq,
    const _Float16* __restrict__ Wkt, const float* __restrict__ bk,
    const _Float16* __restrict__ Wvt, const float* __restrict__ bv,
    _Float16* __restrict__ Qh, _Float16* __restrict__ Kh,
    _Float16* __restrict__ Vt, float* __restrict__ Vf)
{
    const int l  = threadIdx.x;
    const int lm = l & 15, hs = l >> 4;
    const int m0 = blockIdx.x * 16;   // rows over B*N = 4096
    const int n0 = blockIdx.y * 16;   // cols over D = 256
    const int col = n0 + lm;

    v8f aq = {0.f}, ak = {0.f}, av = {0.f};

    for (int kk = 0; kk < 256; kk += 32) {
        // A fragment: Y rows m0..m0+15, k-chunk kk..kk+31 (f32 -> f16)
        const v16h a = loadA_f32(Y + (size_t)(m0 + lm) * 256 + kk + 8 * hs);
        // B fragments: contiguous from transposed f16 weights
        const size_t wb = (size_t)col * 256 + kk + 8 * hs;
        const v16h fq = join8(*(const v8h*)(Wqt + wb), *(const v8h*)(Wqt + wb + 16));
        const v16h fk = join8(*(const v8h*)(Wkt + wb), *(const v8h*)(Wkt + wb + 16));
        const v16h fv = join8(*(const v8h*)(Wvt + wb), *(const v8h*)(Wvt + wb + 16));
        aq = __builtin_amdgcn_wmma_f32_16x16x32_f16(false, a, false, fq, (short)0, aq, false, false);
        ak = __builtin_amdgcn_wmma_f32_16x16x32_f16(false, a, false, fk, (short)0, ak, false, false);
        av = __builtin_amdgcn_wmma_f32_16x16x32_f16(false, a, false, fv, (short)0, av, false, false);
    }

    const float bql = bq[col], bkl = bk[col], bvl = bv[col];
    const int hhead = n0 >> 5;        // head of this 16-col block
    const int din   = col & 31;
#pragma unroll
    for (int r = 0; r < 8; ++r) {
        const int row = m0 + r + 8 * hs;            // D-layout row
        const float q = (aq[r] + bql) * 0.0625f;    // fold 1/sqrt(256)
        const float k = ak[r] + bkl;
        const float v = av[r] + bvl;
        Qh[(size_t)row * 256 + col] = (_Float16)q;
        Kh[(size_t)row * 256 + col] = (_Float16)k;
        Vf[(size_t)row * 256 + col] = v;
        const int bb = row >> 10, nl = row & 1023;
        Vt[((size_t)(bb * 8 + hhead) * 32 + din) * 1024 + nl] = (_Float16)v;
    }
}

// ---------------------------------------------------------------------------
// Kernel 2: fused attention. Block = (b, 16-query tile); 8 waves = 8 heads.
// X_pairs tiles DMA'd into double-buffered LDS by the Tensor Data Mover
// (wave 0 issues; TENSORcnt-synchronized; transfer i+1 overlaps compute i).
// Flash-style online softmax; P transposed via per-wave LDS bounce; WMMA for
// both S = Q K^T and O += P V.
// ---------------------------------------------------------------------------
__global__ __launch_bounds__(256) void attn_kernel(
    const _Float16* __restrict__ Qh, const _Float16* __restrict__ Kh,
    const _Float16* __restrict__ Vt, const float* __restrict__ Xp,
    const float* __restrict__ pres_q, const float* __restrict__ pres_k,
    const float* __restrict__ Wg1, const float* __restrict__ bg1,
    const float* __restrict__ wg2, const float* __restrict__ bg2,
    float* __restrict__ Obuf)
{
    __shared__ alignas(16) float    Xs[2][16 * 96];   // double-buffered DMA tiles
    __shared__ alignas(16) _Float16 Pb[8 * 16 * 32];  // per-wave P transpose buffers

    const int tid = threadIdx.x;
    const int l = tid & 31, w = tid >> 5;   // lane, wave (= head)
    const int lm = l & 15, hs = l >> 4;
    const int n0 = blockIdx.x * 16;
    const int b  = blockIdx.y;

    // per-head geometric MLP weights
    float g[9], b1[3], w2[3];
#pragma unroll
    for (int i = 0; i < 9; ++i) g[i] = Wg1[w * 9 + i];
#pragma unroll
    for (int i = 0; i < 3; ++i) { b1[i] = bg1[w * 3 + i]; w2[i] = wg2[w * 3 + i]; }
    const float b2 = bg2[w];

    // Q A-fragment for this head (dims w*32..w*32+31), loaded once
    const _Float16* qp = Qh + (size_t)(b * 1024 + n0 + lm) * 256 + w * 32;
    const v16h aqf = join8(*(const v8h*)(qp + 8 * hs), *(const v8h*)(qp + 16 + 8 * hs));

    float pq[8];
#pragma unroll
    for (int r = 0; r < 8; ++r) pq[r] = pres_q[b * 1024 + n0 + r + 8 * hs];

    float mrow[8], lrow[8];
#pragma unroll
    for (int r = 0; r < 8; ++r) { mrow[r] = -3.0e38f; lrow[r] = 0.f; }
    v8f o0 = {0.f}, o1 = {0.f};

    _Float16* Pw = Pb + w * 512;

    // X_pairs[b, n0+i, m, c]: rows of 3072 floats; tile = 16 x (32*3) floats
    const float* srcbase = Xp + (size_t)(b * 1024 + n0) * 3072;

    if (w == 0)   // prologue: DMA first tile into buffer 0
        tdm_load_2d_f32(lds_off(&Xs[0][0]), srcbase, 96, 16, 3072, 16, 3072);

    for (int c0 = 0; c0 < 1024; c0 += 32) {
        const int cur = (c0 >> 5) & 1;
        if (w == 0) __builtin_amdgcn_s_wait_tensorcnt(0);   // tile 'cur' landed
        __syncthreads();                                    // visible to all; prev reads done
        if (w == 0 && c0 + 32 < 1024)                       // overlap next DMA with compute
            tdm_load_2d_f32(lds_off(&Xs[cur ^ 1][0]), srcbase + (size_t)(c0 + 32) * 3,
                            96, 16, 3072, 16, 3072);
        const float* Xc = &Xs[cur][0];

        // K B-fragments for key blocks [c0, c0+16), [c0+16, c0+32)
        const _Float16* kp0 = Kh + (size_t)(b * 1024 + c0 + lm) * 256 + w * 32;
        const _Float16* kp1 = kp0 + 16 * 256;
        const v16h bk0 = join8(*(const v8h*)(kp0 + 8 * hs), *(const v8h*)(kp0 + 16 + 8 * hs));
        const v16h bk1 = join8(*(const v8h*)(kp1 + 8 * hs), *(const v8h*)(kp1 + 16 + 8 * hs));
        const v8f zz = {0.f};
        v8f s0 = __builtin_amdgcn_wmma_f32_16x16x32_f16(false, aqf, false, bk0, (short)0, zz, false, false);
        v8f s1 = __builtin_amdgcn_wmma_f32_16x16x32_f16(false, aqf, false, bk1, (short)0, zz, false, false);

        const float pk0 = pres_k[b * 1024 + c0 + lm];
        const float pk1 = pres_k[b * 1024 + c0 + 16 + lm];

#pragma unroll
        for (int r = 0; r < 8; ++r) {
            const int rl = r + 8 * hs;                // local query row of this D element
            float lg0 = s0[r] + locbias(&Xc[rl * 96 + lm * 3],        g, b1, w2, b2);
            float lg1 = s1[r] + locbias(&Xc[rl * 96 + (16 + lm) * 3], g, b1, w2, b2);
            lg0 = pq[r] * lg0 - (1.f - pq[r]) * INF_F;  lg0 = pk0 * lg0 - (1.f - pk0) * INF_F;
            lg1 = pq[r] * lg1 - (1.f - pq[r]) * INF_F;  lg1 = pk1 * lg1 - (1.f - pk1) * INF_F;

            const float rm = halfmax16(fmaxf(lg0, lg1));
            const float mn = fmaxf(mrow[r], rm);
            const float sc = __expf(mrow[r] - mn);
            const float p0 = __expf(lg0 - mn);
            const float p1 = __expf(lg1 - mn);
            const float rs = halfsum16(p0 + p1);
            lrow[r] = lrow[r] * sc + rs;
            mrow[r] = mn;
            o0[r] *= sc; o1[r] *= sc;
            // D-layout -> LDS (per-wave scratch) for A-layout re-read
            Pw[rl * 32 + lm]      = (_Float16)p0;
            Pw[rl * 32 + 16 + lm] = (_Float16)p1;
        }
        __builtin_amdgcn_wave_barrier();   // order LDS write -> read within wave

        // P as A-fragment (16 queries x 32 keys)
        const v16h ap = join8(*(const v8h*)(Pw + lm * 32 + 8 * hs),
                              *(const v8h*)(Pw + lm * 32 + 16 + 8 * hs));
        // V B-fragments from transposed layout: contiguous along keys
        const _Float16* vp0 = Vt + ((size_t)(b * 8 + w) * 32 + lm) * 1024 + c0;
        const _Float16* vp1 = vp0 + 16 * 1024;
        const v16h bv0 = join8(*(const v8h*)(vp0 + 8 * hs), *(const v8h*)(vp0 + 16 + 8 * hs));
        const v16h bv1 = join8(*(const v8h*)(vp1 + 8 * hs), *(const v8h*)(vp1 + 16 + 8 * hs));
        o0 = __builtin_amdgcn_wmma_f32_16x16x32_f16(false, ap, false, bv0, (short)0, o0, false, false);
        o1 = __builtin_amdgcn_wmma_f32_16x16x32_f16(false, ap, false, bv1, (short)0, o1, false, false);
    }

    // normalize and write Oh in head-concatenated [b][n][D] layout
#pragma unroll
    for (int r = 0; r < 8; ++r) {
        const float inv = 1.f / lrow[r];
        const size_t row = (size_t)(b * 1024 + n0 + r + 8 * hs);
        Obuf[row * 256 + w * 32 + lm]      = o0[r] * inv;
        Obuf[row * 256 + w * 32 + 16 + lm] = o1[r] * inv;
    }
}

// ---------------------------------------------------------------------------
// Kernel 3: O1 = V + Oh; out = O1 + relu(O1 @ Wo + bo). One wave per tile.
// ---------------------------------------------------------------------------
__global__ __launch_bounds__(32) void out_kernel(
    const float* __restrict__ Vf, const float* __restrict__ Obuf,
    const _Float16* __restrict__ Wot, const float* __restrict__ bo,
    float* __restrict__ Out)
{
    const int l  = threadIdx.x;
    const int lm = l & 15, hs = l >> 4;
    const int m0 = blockIdx.x * 16, n0 = blockIdx.y * 16;
    const int col = n0 + lm;

    v8f acc = {0.f};
    for (int kk = 0; kk < 256; kk += 32) {
        const size_t base = (size_t)(m0 + lm) * 256 + kk + 8 * hs;
        v16h a;
#pragma unroll
        for (int j = 0; j < 4; ++j) {
            f32x4 v0 = *(const f32x4*)(Vf + base + j * 4 + (j >> 1) * 8);
        }
        // O1 = V + Oh on the fly (vectorized)
        {
            f32x4 va0 = *(const f32x4*)(Vf + base);
            f32x4 va1 = *(const f32x4*)(Vf + base + 4);
            f32x4 vb0 = *(const f32x4*)(Vf + base + 16);
            f32x4 vb1 = *(const f32x4*)(Vf + base + 20);
            f32x4 oa0 = *(const f32x4*)(Obuf + base);
            f32x4 oa1 = *(const f32x4*)(Obuf + base + 4);
            f32x4 ob0 = *(const f32x4*)(Obuf + base + 16);
            f32x4 ob1 = *(const f32x4*)(Obuf + base + 20);
#pragma unroll
            for (int i = 0; i < 4; ++i) {
                a[i]      = (_Float16)(va0[i] + oa0[i]);
                a[i + 4]  = (_Float16)(va1[i] + oa1[i]);
                a[i + 8]  = (_Float16)(vb0[i] + ob0[i]);
                a[i + 12] = (_Float16)(vb1[i] + ob1[i]);
            }
        }
        const size_t wb = (size_t)col * 256 + kk + 8 * hs;
        const v16h f = join8(*(const v8h*)(Wot + wb), *(const v8h*)(Wot + wb + 16));
        acc = __builtin_amdgcn_wmma_f32_16x16x32_f16(false, a, false, f, (short)0, acc, false, false);
    }

    const float bl = bo[col];
#pragma unroll
    for (int r = 0; r < 8; ++r) {
        const size_t row = (size_t)(m0 + r + 8 * hs);
        const float o1 = Vf[row * 256 + col] + Obuf[row * 256 + col];
        Out[row * 256 + col] = o1 + fmaxf(acc[r] + bl, 0.f);
    }
}

// ---------------------------------------------------------------------------
extern "C" void kernel_launch(void* const* d_in, const int* in_sizes, int n_in,
                              void* d_out, int out_size, void* d_ws, size_t ws_size,
                              hipStream_t stream)
{
    (void)in_sizes; (void)n_in; (void)out_size; (void)ws_size;
    // setup_inputs order: 0 X_lift (unused), 1 Y_lift, 2 X_pairs, 3 presence_q,
    // 4 presence_k, 5 Wq, 6 bq, 7 Wk, 8 bk, 9 Wv, 10 bv, 11 Wo, 12 bo,
    // 13 Wg1, 14 bg1, 15 wg2, 16 bg2
    const float* Y   = (const float*)d_in[1];
    const float* Xp  = (const float*)d_in[2];
    const float* pq  = (const float*)d_in[3];
    const float* pk  = (const float*)d_in[4];
    const float* Wq  = (const float*)d_in[5];  const float* bq  = (const float*)d_in[6];
    const float* Wk  = (const float*)d_in[7];  const float* bk  = (const float*)d_in[8];
    const float* Wv  = (const float*)d_in[9];  const float* bv  = (const float*)d_in[10];
    const float* Wo  = (const float*)d_in[11]; const float* bo  = (const float*)d_in[12];
    const float* Wg1 = (const float*)d_in[13]; const float* bg1 = (const float*)d_in[14];
    const float* wg2 = (const float*)d_in[15]; const float* bg2 = (const float*)d_in[16];

    // workspace layout (14.5 MB total)
    char* ws = (char*)d_ws;
    _Float16* Qh  = (_Float16*)(ws + (size_t)0);           // 2 MB  [B*N][D] f16 (pre-scaled)
    _Float16* Kh  = (_Float16*)(ws + ((size_t)2  << 20));  // 2 MB  [B*N][D] f16
    _Float16* Vt  = (_Float16*)(ws + ((size_t)4  << 20));  // 2 MB  [B][H][DH][N] f16
    float*    Vf  = (float*)   (ws + ((size_t)6  << 20));  // 4 MB  [B*N][D] f32
    float*    Ob  = (float*)   (ws + ((size_t)10 << 20));  // 4 MB  [B*N][D] f32
    _Float16* Wqt = (_Float16*)(ws + ((size_t)14 << 20));              // 128 KB each
    _Float16* Wkt = (_Float16*)(ws + ((size_t)14 << 20) + (128 << 10));
    _Float16* Wvt = (_Float16*)(ws + ((size_t)14 << 20) + (256 << 10));
    _Float16* Wot = (_Float16*)(ws + ((size_t)14 << 20) + (384 << 10));

    wprep_kernel<<<dim3(256), 256, 0, stream>>>(Wq, Wk, Wv, Wo, Wqt, Wkt, Wvt, Wot);
    qkv_kernel<<<dim3(256, 16), 32, 0, stream>>>(Y, Wqt, bq, Wkt, bk, Wvt, bv, Qh, Kh, Vt, Vf);
    attn_kernel<<<dim3(64, 4), 256, 0, stream>>>(Qh, Kh, Vt, Xp, pq, pk,
                                                 Wg1, bg1, wg2, bg2, Ob);
    out_kernel<<<dim3(256, 16), 32, 0, stream>>>(Vf, Ob, Wot, bo, (float*)d_out);
}
